// GraphTransformerLayer_47167330845010
// MI455X (gfx1250) — matrix operand order repaired and checked by reading, not compiled, not run on hardware
//
#include <hip/hip_runtime.h>
#include <hip/hip_bf16.h>

typedef __attribute__((ext_vector_type(16))) _Float16 v16h;
typedef __attribute__((ext_vector_type(8)))  _Float16 v8h;
typedef __attribute__((ext_vector_type(8)))  float    v8f;

#define GTN 4096
#define GTD 256
#define GTH 8
#define GHD 32
#define LOG2E 1.44269504088896340736f
#define SM_SCALE 0.0625f                 /* 1/sqrt(256) */
#define QSCALE (SM_SCALE * LOG2E)        /* folded into Q so S is in exp2 domain */
#define LEAKY 0.2f

// ---------------------------------------------------------------------------
// WMMA tile loaders (16x16x32 f16, wave32; contiguous k order used for both A and B).
// A (16x32, f16, row-major src): row m = lane&15;
//   elem e -> K = (lane>=16?16:0)+e
// B (32x16, f16, src with K contiguous per lane): col n = lane&15;
//   elem e -> K = (lane>=16?16:0)+e
// C (16x16, f32): vgpr j -> row j+(lane>=16?8:0); col = lane&15
// ---------------------------------------------------------------------------
static __device__ __forceinline__ v16h load_a_f16(const _Float16* __restrict__ base,
                                                  int ld, int lane) {
  const int m  = lane & 15;
  const int kb = (lane & 16) ? 16 : 0;
  const _Float16* p = base + (size_t)m * ld + kb;
  v16h a;
#pragma unroll
  for (int e = 0; e < 8; ++e) a[e] = p[e];
#pragma unroll
  for (int e = 0; e < 8; ++e) a[8 + e] = p[8 + e];
  return a;
}

// base points at Xt[col0 * ld + k0] of a transposed ([N][K] row-major) matrix
static __device__ __forceinline__ v16h load_bt_f16(const _Float16* __restrict__ base,
                                                   int ld, int lane) {
  const int n  = lane & 15;
  const int kb = (lane & 16) ? 16 : 0;
  const _Float16* p = base + (size_t)n * ld + kb;
  v16h b;
#pragma unroll
  for (int e = 0; e < 16; ++e) b[e] = p[e];
  return b;
}

static __device__ __forceinline__ v8f hwmma(v16h a, v16h b, v8f c) {
  return __builtin_amdgcn_wmma_f32_16x16x32_f16(false, a, false, b,
                                                (short)0, c, false, false);
}

// ---------------------------------------------------------------------------
// Kernel 0: convert X -> f16; weights -> TRANSPOSED f16; zero K2 accumulator
// ---------------------------------------------------------------------------
__global__ void gt_prep_kernel(const float* __restrict__ X,
                               const float* __restrict__ WQ,
                               const float* __restrict__ WK,
                               const float* __restrict__ WV,
                               const float* __restrict__ Wc,
                               _Float16* __restrict__ Xh,
                               _Float16* __restrict__ WQt,
                               _Float16* __restrict__ WKt,
                               _Float16* __restrict__ WVt,
                               _Float16* __restrict__ Wct,
                               float* __restrict__ K2f) {
  const int i = blockIdx.x * blockDim.x + threadIdx.x;
  if (i < GTN * GTD) {
    Xh[i]  = (_Float16)X[i];
    K2f[i] = 0.0f;
  }
  if (i < GTD * GTD) {
    const int k = i >> 8;          // source row
    const int n = i & 255;         // source col
    const int it = n * GTD + k;    // transposed index
    WQt[it] = (_Float16)WQ[i];
    WKt[it] = (_Float16)WK[i];
    WVt[it] = (_Float16)WV[i];
    Wct[it] = (_Float16)Wc[i];
  }
}

// ---------------------------------------------------------------------------
// Kernel 1: fused QKV GEMM.  One wave -> 16x64 output tile of Q, K or V.
// Q -> f16 row-major, PRE-SCALED by SM_SCALE*log2(e) (softmax exp2 domain),
// K -> f32 row-major (for atomic scatter),
// V -> f16 TRANSPOSED [headdim][node] (A-operand layout for O^T = V^T P^T).
// ---------------------------------------------------------------------------
__global__ void __launch_bounds__(32)
gt_qkv_gemm_kernel(const _Float16* __restrict__ Xh,
                   const _Float16* __restrict__ WQt,
                   const _Float16* __restrict__ WKt,
                   const _Float16* __restrict__ WVt,
                   _Float16* __restrict__ Qh,
                   float*    __restrict__ Kf,
                   _Float16* __restrict__ Vt) {
  const int lane = threadIdx.x & 31;
  const int r0 = blockIdx.x * 16;
  const int c0 = blockIdx.y * 64;
  const int z  = blockIdx.z;
  const _Float16* __restrict__ Wt = (z == 0) ? WQt : (z == 1) ? WKt : WVt;

  v8f acc[4] = {};
  for (int k0 = 0; k0 < GTD; k0 += 32) {
    const v16h a = load_a_f16(Xh + (size_t)r0 * GTD + k0, GTD, lane);
#pragma unroll
    for (int t = 0; t < 4; ++t) {
      const v16h b = load_bt_f16(Wt + (size_t)(c0 + 16 * t) * GTD + k0, GTD, lane);
      acc[t] = hwmma(a, b, acc[t]);
    }
  }

  const int m  = lane & 15;
  const int rr = (lane & 16) ? 8 : 0;
  if (z == 0) {
#pragma unroll
    for (int t = 0; t < 4; ++t)
#pragma unroll
      for (int j = 0; j < 8; ++j)
        Qh[(size_t)(r0 + rr + j) * GTD + c0 + 16 * t + m] =
            (_Float16)(acc[t][j] * QSCALE);
  } else if (z == 1) {
#pragma unroll
    for (int t = 0; t < 4; ++t)
#pragma unroll
      for (int j = 0; j < 8; ++j)
        Kf[(size_t)(r0 + rr + j) * GTD + c0 + 16 * t + m] = acc[t][j];
  } else {
    // transposed store: VT[col][row]; 8 contiguous halves per lane -> b128
#pragma unroll
    for (int t = 0; t < 4; ++t) {
      v8h pv;
#pragma unroll
      for (int j = 0; j < 8; ++j) pv[j] = (_Float16)acc[t][j];
      *(v8h*)(Vt + (size_t)(c0 + 16 * t + m) * GTN + r0 + rr) = pv;
    }
  }
}

// ---------------------------------------------------------------------------
// Kernel 2: K2 = segment_sum(K[src], dst).  32 lanes per edge, 8 floats/lane.
// ---------------------------------------------------------------------------
__global__ void gt_segsum_kernel(const int* __restrict__ src,
                                 const int* __restrict__ dst,
                                 const float* __restrict__ Kf,
                                 float* __restrict__ K2f, int E) {
  const int t = blockIdx.x * blockDim.x + threadIdx.x;
  const int e = t >> 5;
  if (e >= E) return;
  const int c0 = (t & 31) * 8;
  const int s = src[e];
  const int d = dst[e];
  const float* __restrict__ kp = Kf + (size_t)s * GTD + c0;
  float* __restrict__ op = K2f + (size_t)d * GTD + c0;
#pragma unroll
  for (int i = 0; i < 8; ++i) atomicAdd(op + i, kp[i]);
}

// Kernel 3: K2 f32 -> f16
__global__ void gt_k2cvt_kernel(const float* __restrict__ K2f,
                                _Float16* __restrict__ K2h) {
  const int i = blockIdx.x * blockDim.x + threadIdx.x;
  if (i < GTN * GTD) K2h[i] = (_Float16)K2f[i];
}

// ---------------------------------------------------------------------------
// Kernel 4: streaming (flash) attention, TRANSPOSED.  4 waves/block; each
// wave owns 16 queries of one head and sweeps the 4096 keys in 32-wide chunks.
//   S^T = K2 @ Q^T  (2 WMMAs)  -> each lane holds 16 of its query's scores
//   online softmax: in-lane max/sum trees + one shfl_xor(16) each
//   P^T -> B layout via 2 packed ds_store_b128 + 2 ds_load_b128
//   O^T += V^T @ P^T (2 WMMAs) using the pre-transposed Vt
// Per-query softmax state (m, l) is one scalar per lane.
// ---------------------------------------------------------------------------
__global__ void __launch_bounds__(128)
gt_attn_kernel(const _Float16* __restrict__ Qh,   // pre-scaled by SM_SCALE*log2e
               const _Float16* __restrict__ K2h,
               const _Float16* __restrict__ Vt,
               float* __restrict__ attnf) {
  __shared__ alignas(16) _Float16 pbuf[4][16][40];   // [wave][query][key(+pad)]

  const int lane = threadIdx.x & 31;
  const int wid  = threadIdx.x >> 5;
  const int h    = blockIdx.y;
  const int r0   = (blockIdx.x * 4 + wid) * 16;
  const int m    = lane & 15;
  const int kbB  = (lane & 16) ? 16 : 0;
  const int rr   = (lane & 16) ? 8 : 0;

  // B operand for S^T: lane n = query, elems = head dims (contiguous)
  const _Float16* __restrict__ qp = Qh + (size_t)(r0 + m) * GTD + h * GHD + kbB;
  v16h bq;
#pragma unroll
  for (int e = 0; e < 16; ++e) bq[e] = qp[e];

  v8f o0 = {}, o1 = {};        // O^T accumulators: rows = head dims, col = query
  const v8f czero = {};
  float mq = -3.0e38f, lq = 0.0f;   // per-lane (per-query) softmax state

  const _Float16* __restrict__ vrow0 = Vt + (size_t)(h * GHD) * GTN;
  const _Float16* __restrict__ vrow1 = Vt + (size_t)(h * GHD + 16) * GTN;
  _Float16* prow = &pbuf[wid][m][0];

  for (int n0 = 0; n0 < GTN; n0 += 32) {
    // --- S^T tiles: A = K2 rows (keys x headdim), B = Q^T ---
    const v16h ak0 = load_a_f16(K2h + (size_t)n0 * GTD + h * GHD, GTD, lane);
    const v16h ak1 = load_a_f16(K2h + (size_t)(n0 + 16) * GTD + h * GHD, GTD, lane);
    const v8f st0 = hwmma(ak0, bq, czero);   // keys n0+0..15   (rows), queries (cols)
    const v8f st1 = hwmma(ak1, bq, czero);   // keys n0+16..31

    // --- per-lane online softmax over the 32-key chunk ---
    float cm = fmaxf(st0[0], st1[0]);
#pragma unroll
    for (int j = 1; j < 8; ++j) cm = fmaxf(cm, fmaxf(st0[j], st1[j]));
    cm = fmaxf(cm, __shfl_xor(cm, 16, 32));          // merge the query's two halves
    const float mn   = fmaxf(mq, cm);
    const float corr = exp2f(mq - mn);
    v8h pk0, pk1;
    float rs = 0.0f;
#pragma unroll
    for (int j = 0; j < 8; ++j) {
      const float e0 = exp2f(st0[j] - mn);
      const float e1 = exp2f(st1[j] - mn);
      rs += e0 + e1;
      pk0[j] = (_Float16)e0;
      pk1[j] = (_Float16)e1;
    }
    rs += __shfl_xor(rs, 16, 32);
    lq = lq * corr + rs;
    mq = mn;
#pragma unroll
    for (int j = 0; j < 8; ++j) { o0[j] *= corr; o1[j] *= corr; }

    // --- P^T (C layout) -> B layout via LDS: pbuf[query][key] ---
    *(v8h*)(prow + rr)      = pk0;    // keys rr..rr+7
    *(v8h*)(prow + 16 + rr) = pk1;    // keys 16+rr..16+rr+7
    asm volatile("s_wait_dscnt 0" ::: "memory");      // same-wave LDS RAW fence
    v16h bp;
#pragma unroll
    for (int e = 0; e < 16; ++e) bp[e] = prow[kbB + e];

    // --- O^T += V^T @ P^T : A tiles contiguous along node axis in Vt ---
    const v16h av0 = load_a_f16(vrow0 + n0, GTN, lane);
    const v16h av1 = load_a_f16(vrow1 + n0, GTN, lane);
    o0 = hwmma(av0, bp, o0);
    o1 = hwmma(av1, bp, o1);
  }

  // finalize: one reciprocal per query-lane; contiguous per-lane stores
  const float inv = 1.0f / lq;
  float* __restrict__ op = attnf + (size_t)(r0 + m) * GTD + h * GHD + rr;
#pragma unroll
  for (int j = 0; j < 8; ++j) op[j] = o0[j] * inv;       // head dims rr..rr+7
#pragma unroll
  for (int j = 0; j < 8; ++j) op[16 + j] = o1[j] * inv;  // head dims 16+rr..+7
}

// Kernel 5: h = X + attn (keep f32 for residual, f16 for the WMMA GEMM)
__global__ void gt_residual_kernel(const float* __restrict__ X,
                                   const float* __restrict__ attnf,
                                   float* __restrict__ hf,
                                   _Float16* __restrict__ hh) {
  const int i = blockIdx.x * blockDim.x + threadIdx.x;
  if (i < GTN * GTD) {
    const float v = X[i] + attnf[i];
    hf[i] = v;
    hh[i] = (_Float16)v;
  }
}

// ---------------------------------------------------------------------------
// Kernel 6: out = h + leaky_relu(h @ Wc + bc).  One wave -> 16x64 tile.
// ---------------------------------------------------------------------------
__global__ void __launch_bounds__(32)
gt_out_gemm_kernel(const _Float16* __restrict__ hh,
                   const _Float16* __restrict__ Wct,
                   const float* __restrict__ bc,
                   const float* __restrict__ hf,
                   float* __restrict__ out) {
  const int lane = threadIdx.x & 31;
  const int r0 = blockIdx.x * 16;
  const int c0 = blockIdx.y * 64;

  v8f acc[4] = {};
  for (int k0 = 0; k0 < GTD; k0 += 32) {
    const v16h a = load_a_f16(hh + (size_t)r0 * GTD + k0, GTD, lane);
#pragma unroll
    for (int t = 0; t < 4; ++t) {
      const v16h b = load_bt_f16(Wct + (size_t)(c0 + 16 * t) * GTD + k0, GTD, lane);
      acc[t] = hwmma(a, b, acc[t]);
    }
  }

  const int m  = lane & 15;
  const int rr = (lane & 16) ? 8 : 0;
#pragma unroll
  for (int t = 0; t < 4; ++t) {
#pragma unroll
    for (int j = 0; j < 8; ++j) {
      const int row = r0 + rr + j;
      const int col = c0 + 16 * t + m;
      float v = acc[t][j] + bc[col];
      v = (v > 0.0f) ? v : LEAKY * v;
      const size_t idx = (size_t)row * GTD + col;
      out[idx] = hf[idx] + v;
    }
  }
}

// ---------------------------------------------------------------------------
// Host launcher
// ---------------------------------------------------------------------------
extern "C" void kernel_launch(void* const* d_in, const int* in_sizes, int n_in,
                              void* d_out, int out_size, void* d_ws, size_t ws_size,
                              hipStream_t stream) {
  (void)n_in; (void)out_size; (void)ws_size;
  const float* X   = (const float*)d_in[0];
  const int*   src = (const int*)d_in[1];
  const int*   dst = (const int*)d_in[2];
  const float* WQ  = (const float*)d_in[3];
  const float* WK  = (const float*)d_in[4];
  const float* WV  = (const float*)d_in[5];
  const float* Wc  = (const float*)d_in[6];
  const float* bc  = (const float*)d_in[7];
  const int E = in_sizes[1];

  char* w = (char*)d_ws;
  auto carve = [&](size_t bytes) -> char* {
    char* p = w;
    w += (bytes + 255) & ~(size_t)255;
    return p;
  };
  const size_t ND = (size_t)GTN * GTD;
  const size_t DD = (size_t)GTD * GTD;
  _Float16* Xh    = (_Float16*)carve(ND * 2);
  _Float16* Qh    = (_Float16*)carve(ND * 2);
  float*    Kf    = (float*)   carve(ND * 4);
  _Float16* Vt    = (_Float16*)carve(ND * 2);   // transposed [headdim][node]
  float*    K2f   = (float*)   carve(ND * 4);
  _Float16* K2h   = (_Float16*)carve(ND * 2);
  float*    attnf = (float*)   carve(ND * 4);
  float*    hf    = (float*)   carve(ND * 4);
  _Float16* hh    = (_Float16*)carve(ND * 2);
  _Float16* WQt   = (_Float16*)carve(DD * 2);
  _Float16* WKt   = (_Float16*)carve(DD * 2);
  _Float16* WVt   = (_Float16*)carve(DD * 2);
  _Float16* Wct   = (_Float16*)carve(DD * 2);

  const int nd = GTN * GTD;

  gt_prep_kernel<<<nd / 256, 256, 0, stream>>>(X, WQ, WK, WV, Wc,
                                               Xh, WQt, WKt, WVt, Wct, K2f);

  gt_qkv_gemm_kernel<<<dim3(GTN / 16, GTD / 64, 3), 32, 0, stream>>>(
      Xh, WQt, WKt, WVt, Qh, Kf, Vt);

  {
    const long long tot = (long long)E * 32;
    const int blocks = (int)((tot + 255) / 256);
    gt_segsum_kernel<<<blocks, 256, 0, stream>>>(src, dst, Kf, K2f, E);
  }

  gt_k2cvt_kernel<<<nd / 256, 256, 0, stream>>>(K2f, K2h);

  gt_attn_kernel<<<dim3(GTN / 16 / 4, GTH), 128, 0, stream>>>(Qh, K2h, Vt, attnf);

  gt_residual_kernel<<<nd / 256, 256, 0, stream>>>(X, attnf, hf, hh);

  gt_out_gemm_kernel<<<dim3(GTN / 16, GTD / 64), 32, 0, stream>>>(
      hh, Wct, bc, hf, (float*)d_out);
}
